// RankBoost_39651138077168
// MI455X (gfx1250) — matrix operand order repaired and checked by reading, not compiled, not run on hardware
//
#include <hip/hip_runtime.h>

// ---------------------------------------------------------------------------
// RankBoost fused implementation for gfx1250 (MI455X, wave32, WMMA).
//
// Phase plan (all on `stream`, stream-ordered):
//   k_prep   : f32 -> f16 weight conversions (+transposes) into workspace
//   k_lstm   : 1 workgroup, 8 waves, WMMA LSTM over 50 steps (state in VGPR/LDS)
//   k_aprime : a'[l] = seq_repr @ W1[l,:d] + b1[l]   (tiny, VALU)
//   k_score  : 625 blocks x 256 thr; per 16-item tile: WMMA bterm, then fused
//              relu(a'+bterm) -> WMMA @ W2 -> relu -> dot W3 -> scores.
//              W2^T staged to LDS via GLOBAL_LOAD_ASYNC_TO_LDS_B128 (ASYNCcnt).
//   k_loss   : pos/neg gather + mean(exp(-(pos-neg)))
// ---------------------------------------------------------------------------

typedef _Float16 h8   __attribute__((ext_vector_type(8)));
typedef _Float16 v16h __attribute__((ext_vector_type(16)));
typedef float    v8f  __attribute__((ext_vector_type(8)));
typedef int      v4i  __attribute__((vector_size(16)));

typedef v4i __attribute__((address_space(1)))* gv4i_p;  // global (AS1)
typedef v4i __attribute__((address_space(3)))* lv4i_p;  // LDS (AS3)

#define DIMD   128
#define NIT    10000
#define NL     4
#define BATCH  64
#define SEQT   50

// workspace byte offsets (ws is >=256B aligned)
#define OFF_SEQR   0u                       // 64*128 f32      = 32768
#define OFF_AP     32768u                   // 4*64*128 f32    = 131072
#define OFF_WIH    163840u                  // 512*128 f16     = 131072
#define OFF_WHH    294912u                  // 512*128 f16     = 131072
#define OFF_W1BT   425984u                  // 4*128*128 f16   = 131072
#define OFF_W2T    557056u                  // 4*64*128 f16    = 65536

#define HAS_ASYNC_LDS __has_builtin(__builtin_amdgcn_global_load_async_to_lds_b128)

static __device__ __forceinline__ v16h cat16(h8 lo, h8 hi) {
  return __builtin_shufflevector(lo, hi, 0,1,2,3,4,5,6,7,8,9,10,11,12,13,14,15);
}
static __device__ __forceinline__ float sigmoidf_(float x) {
  return 1.0f / (1.0f + __expf(-x));
}
static __device__ __forceinline__ v8f wmma_f16(v16h a, v16h b, v8f c) {
  return __builtin_amdgcn_wmma_f32_16x16x32_f16(false, a, false, b, (short)0, c,
                                                false, false);
}

// ---------------------------------------------------------------------------
// k_prep: f16 conversions.
//   wih[g][k]  = W_ih[g,k]            (B-layout friendly: col g of W_ih^T)
//   whh[g][k]  = W_hh[g,k]
//   w1bt[l][n][k] = W1[l, 128+k, n]   (B cols of W1[d:])
//   w2t [l][h][k] = W2[l, k, h]       (B cols of W2)
// ---------------------------------------------------------------------------
__global__ void k_prep(const float* __restrict__ W_ih,
                       const float* __restrict__ W_hh,
                       const float* __restrict__ W1,
                       const float* __restrict__ W2,
                       char* __restrict__ ws) {
  _Float16* wih  = (_Float16*)(ws + OFF_WIH);
  _Float16* whh  = (_Float16*)(ws + OFF_WHH);
  _Float16* w1bt = (_Float16*)(ws + OFF_W1BT);
  _Float16* w2t  = (_Float16*)(ws + OFF_W2T);
  int tid = blockIdx.x * blockDim.x + threadIdx.x;
  int stride = gridDim.x * blockDim.x;
  for (int i = tid; i < 512 * 128; i += stride) {
    wih[i] = (_Float16)W_ih[i];
    whh[i] = (_Float16)W_hh[i];
  }
  for (int i = tid; i < NL * 128 * 128; i += stride) {
    int l = i >> 14, r = i & 16383, n = r >> 7, k = r & 127;
    w1bt[i] = (_Float16)W1[l * 32768 + (128 + k) * 128 + n];
  }
  for (int i = tid; i < NL * 64 * 128; i += stride) {
    int l = i >> 13, r = i & 8191, hh = r >> 7, k = r & 127;
    w2t[i] = (_Float16)W2[l * 8192 + k * 64 + hh];
  }
}

// ---------------------------------------------------------------------------
// k_lstm: one workgroup (8 waves). Per step:
//   gates(64x512) = A(64x256=[X|H]) @ B(256x512=[Wih^T;Whh^T])  (f16 WMMA)
// Wave w owns hidden-column tile dd=16w..16w+15 for all 4 batch tiles;
// cell state lives in VGPRs across steps. H is re-written as f16 into the
// shared A buffer for the next step.
// ---------------------------------------------------------------------------
__global__ void __launch_bounds__(256) k_lstm(const int* __restrict__ user_seq,
                                              const float* __restrict__ emb,
                                              const float* __restrict__ b_ih,
                                              const float* __restrict__ b_hh,
                                              char* __restrict__ ws) {
  __shared__ __align__(16) _Float16 Abuf[64 * 256];  // row b, K 0..255
  __shared__ float bias[512];
  const _Float16* wih = (const _Float16*)(ws + OFF_WIH);
  const _Float16* whh = (const _Float16*)(ws + OFF_WHH);
  float* seq_repr = (float*)(ws + OFF_SEQR);

  const int tid = threadIdx.x;
  const int wave = tid >> 5, lane = tid & 31;
  const int r16 = lane & 15;
  const bool hiH = lane >= 16;
  const int dd = wave * 16 + r16;

  for (int i = tid; i < 512; i += 256) bias[i] = b_ih[i] + b_hh[i];
  for (int i = tid; i < 64 * 128; i += 256)       // zero H region (h0 = 0)
    Abuf[(i >> 7) * 256 + 128 + (i & 127)] = (_Float16)0.0f;

  float c_reg[4][8];
#pragma unroll
  for (int mt = 0; mt < 4; mt++)
#pragma unroll
    for (int rr = 0; rr < 8; rr++) c_reg[mt][rr] = 0.0f;

  for (int t = 0; t < SEQT; t++) {
    // gather x_t: thread handles (b = tid/4, 32 K values)
    {
      int b = tid >> 2;
      int kbase = (tid & 3) * 32;
      int item = user_seq[b * SEQT + t];
      const float* src = emb + item * DIMD + kbase;
      _Float16* dst = &Abuf[b * 256 + kbase];
#pragma unroll
      for (int k = 0; k < 32; k++) dst[k] = (_Float16)src[k];
    }
    __syncthreads();  // S1: A ([X_t | H_{t-1}]) ready

    float hnew[4][8];
#pragma unroll
    for (int mt = 0; mt < 4; mt++) {
      v8f acc[4] = {};
#pragma unroll
      for (int kk = 0; kk < 8; kk++) {
        const int brow = mt * 16 + r16;
        const int kof = kk * 32 + (hiH ? 8 : 0);
        const h8* pa = (const h8*)&Abuf[brow * 256 + kof];
        const h8* pa2 = (const h8*)&Abuf[brow * 256 + kof + 16];
        v16h afrag = cat16(*pa, *pa2);
#pragma unroll
        for (int q = 0; q < 4; q++) {
          const int ncol = q * 128 + dd;
          const _Float16* wb = (kk < 4) ? &wih[ncol * 128 + kk * 32]
                                        : &whh[ncol * 128 + (kk - 4) * 32];
          const h8* pb = (const h8*)(wb + (hiH ? 8 : 0));
          v16h bfrag = cat16(pb[0], pb[2]);
          acc[q] = wmma_f16(afrag, bfrag, acc[q]);
        }
      }
#pragma unroll
      for (int rr = 0; rr < 8; rr++) {
        float iv = sigmoidf_(acc[0][rr] + bias[dd]);
        float fv = sigmoidf_(acc[1][rr] + bias[128 + dd]);
        float gv = tanhf(acc[2][rr] + bias[256 + dd]);
        float ov = sigmoidf_(acc[3][rr] + bias[384 + dd]);
        float cn = fv * c_reg[mt][rr] + iv * gv;
        c_reg[mt][rr] = cn;
        hnew[mt][rr] = ov * tanhf(cn);
      }
    }
    __syncthreads();  // S2: all A reads retired -> safe to overwrite H

#pragma unroll
    for (int mt = 0; mt < 4; mt++)
#pragma unroll
      for (int rr = 0; rr < 8; rr++) {
        int brow = mt * 16 + rr + (hiH ? 8 : 0);
        Abuf[brow * 256 + 128 + dd] = (_Float16)hnew[mt][rr];
        if (t == SEQT - 1) seq_repr[brow * DIMD + dd] = hnew[mt][rr];
      }
  }
}

// ---------------------------------------------------------------------------
// k_aprime: a'[l][b][n] = sum_k seq_repr[b][k] * W1[l,k,n] + b1[l,n]
// ---------------------------------------------------------------------------
__global__ void k_aprime(const float* __restrict__ W1,
                         const float* __restrict__ b1,
                         char* __restrict__ ws) {
  int l = blockIdx.x;
  const float* sr = (const float*)(ws + OFF_SEQR);
  float* ap = (float*)(ws + OFF_AP) + l * BATCH * DIMD;
  const float* w = W1 + l * 256 * 128;  // rows 0..127 = W1[l,:d]
  for (int e = threadIdx.x; e < BATCH * DIMD; e += blockDim.x) {
    int b = e >> 7, n = e & 127;
    float s = b1[l * DIMD + n];
    for (int k = 0; k < DIMD; k++) s += sr[b * DIMD + k] * w[k * 128 + n];
    ap[e] = s;
  }
}

// ---------------------------------------------------------------------------
// k_score: one block per 16-item tile (625 blocks), 8 waves.
// Per learner l:
//   stage W2^T (f16) into LDS (async-to-LDS when available); wave w computes
//   bterm col-tile w via WMMA; then each wave runs 8 batches:
//   A=relu(a'+bterm) -> 16 WMMA vs W2 -> relu(+b2) dot W3 via width-16
//   shuffle reduce -> scores accum in LDS.
// Final: plain store of 64x16 score patch (+ sum_l alpha_l*b3_l).
// ---------------------------------------------------------------------------
__global__ void __launch_bounds__(256) k_score(const float* __restrict__ emb,
                                               const float* __restrict__ W3,
                                               const float* __restrict__ b2,
                                               const float* __restrict__ b3,
                                               const float* __restrict__ alphas,
                                               const char* __restrict__ ws,
                                               float* __restrict__ out) {
  __shared__ __align__(16) float bterm[16 * 128];      // 8 KB
  __shared__ __align__(16) _Float16 w2t_l[64 * 128];   // 16 KB
  __shared__ float scores[64 * 16];                    // 4 KB

  const int nt = blockIdx.x;
  const int tid = threadIdx.x, wave = tid >> 5, lane = tid & 31;
  const int r16 = lane & 15;
  const bool hiH = lane >= 16;

  const _Float16* w1bt = (const _Float16*)(ws + OFF_W1BT);
  const _Float16* w2t = (const _Float16*)(ws + OFF_W2T);
  const float* ap = (const float*)(ws + OFF_AP);

  for (int i = tid; i < 64 * 16; i += 256) scores[i] = 0.0f;

  const int item = nt * 16 + r16;
  __builtin_prefetch(emb + item * DIMD, 0, 0);  // global_prefetch_b8

  for (int l = 0; l < NL; l++) {
    __syncthreads();  // prev-iteration readers of bterm / w2t_l are done

    // ---- stage W2^T[l] (16 KB f16) into LDS ----
#if HAS_ASYNC_LDS
    {
      // Per-lane 16B global->LDS DMA, tracked by ASYNCcnt (no VGPR data).
      gv4i_p g = (gv4i_p)(w2t + l * 8192) + tid;
      lv4i_p s = (lv4i_p)(&w2t_l[0]) + tid;
#pragma unroll
      for (int it = 0; it < 4; it++)
        __builtin_amdgcn_global_load_async_to_lds_b128(g + it * 256,
                                                       s + it * 256, 0, 0);
    }
#else
    for (int i = tid; i < 64 * 128; i += 256) w2t_l[i] = w2t[l * 8192 + i];
#endif

    // bterm col-tile `wave` : (16 items x 128) @ W1[l,d:]
    {
      v8f acc = {};
#pragma unroll
      for (int kk = 0; kk < 4; kk++) {
        const int kof = kk * 32 + (hiH ? 8 : 0);
        const float* pe = emb + item * DIMD + kof;
        v16h afrag;
#pragma unroll
        for (int e = 0; e < 8; e++) {
          afrag[e] = (_Float16)pe[e];
          afrag[8 + e] = (_Float16)pe[16 + e];
        }
        const int ncol = wave * 16 + r16;
        const h8* pb =
            (const h8*)(w1bt + l * 16384 + ncol * 128 + kk * 32 + (hiH ? 8 : 0));
        v16h bfrag = cat16(pb[0], pb[2]);
        acc = wmma_f16(afrag, bfrag, acc);
      }
#pragma unroll
      for (int rr = 0; rr < 8; rr++) {
        int m = rr + (hiH ? 8 : 0);
        bterm[m * 128 + wave * 16 + r16] = acc[rr];
      }
    }

#if HAS_ASYNC_LDS
#if __has_builtin(__builtin_amdgcn_s_wait_asynccnt)
    __builtin_amdgcn_s_wait_asynccnt(0);
#else
    asm volatile("s_wait_asynccnt 0" ::: "memory");
#endif
#endif
    __syncthreads();  // bterm + staged W2^T visible

    const float alpha = alphas[l];
    float w3v[4], b2v[4];
#pragma unroll
    for (int nj = 0; nj < 4; nj++) {
      int hcol = nj * 16 + r16;
      w3v[nj] = W3[l * 64 + hcol];
      b2v[nj] = b2[l * 64 + hcol];
    }

    for (int bb = 0; bb < 8; bb++) {  // wave handles b = wave*8 .. wave*8+7
      const int b = wave * 8 + bb;
      const float* apb = ap + l * BATCH * DIMD + b * DIMD;
      v8f acc[4] = {};
#pragma unroll
      for (int kk = 0; kk < 4; kk++) {
        const int k0 = kk * 32 + (hiH ? 8 : 0);
        const int k1 = k0 + 16;
        const float* bt = &bterm[r16 * 128];
        v16h afrag;
#pragma unroll
        for (int e = 0; e < 8; e++) {
          afrag[e] = (_Float16)fmaxf(apb[k0 + e] + bt[k0 + e], 0.0f);
          afrag[8 + e] = (_Float16)fmaxf(apb[k1 + e] + bt[k1 + e], 0.0f);
        }
#pragma unroll
        for (int nj = 0; nj < 4; nj++) {
          const h8* pb =
              (const h8*)&w2t_l[(nj * 16 + r16) * 128 + kk * 32 + (hiH ? 8 : 0)];
          v16h bfrag = cat16(pb[0], pb[2]);
          acc[nj] = wmma_f16(afrag, bfrag, acc[nj]);
        }
      }
      // score[row] = sum_h relu(h2 + b2) * W3[h]; reduce over 16 lanes
      float myv = 0.0f;
#pragma unroll
      for (int rr = 0; rr < 8; rr++) {
        float s = 0.0f;
#pragma unroll
        for (int nj = 0; nj < 4; nj++)
          s += fmaxf(acc[nj][rr] + b2v[nj], 0.0f) * w3v[nj];
        s += __shfl_xor(s, 1, 16);
        s += __shfl_xor(s, 2, 16);
        s += __shfl_xor(s, 4, 16);
        s += __shfl_xor(s, 8, 16);
        if (r16 == rr) myv = s;  // lane r16 keeps row r16 (cndmask, no spill)
      }
      if (r16 < 8) {
        int row = r16 + (hiH ? 8 : 0);
        scores[b * 16 + row] += alpha * myv;
      }
    }
  }
  __syncthreads();

  float cb3 = 0.0f;
  for (int l = 0; l < NL; l++) cb3 += alphas[l] * b3[l];
  for (int i = tid; i < 64 * 16; i += 256) {
    int b = i >> 4, it = i & 15;
    out[b * NIT + nt * 16 + it] = scores[i] + cb3;
  }
}

// ---------------------------------------------------------------------------
// k_loss: loss = mean(exp(-(pos - neg)))
// ---------------------------------------------------------------------------
__global__ void k_loss(const int* __restrict__ pos, const int* __restrict__ neg,
                       float* __restrict__ out) {
  __shared__ float red[64];
  int b = threadIdx.x;
  float p = out[b * NIT + pos[b]];
  float n = out[b * NIT + neg[b]];
  red[b] = __expf(-(p - n));
  __syncthreads();
  if (b == 0) {
    float s = 0.0f;
    for (int i = 0; i < 64; i++) s += red[i];
    out[BATCH * NIT] = s / 64.0f;
  }
}

extern "C" void kernel_launch(void* const* d_in, const int* in_sizes, int n_in,
                              void* d_out, int out_size, void* d_ws,
                              size_t ws_size, hipStream_t stream) {
  const int* user_seq = (const int*)d_in[0];
  const int* pos_item = (const int*)d_in[1];
  const int* neg_item = (const int*)d_in[2];
  const float* emb    = (const float*)d_in[3];
  const float* W_ih   = (const float*)d_in[4];
  const float* W_hh   = (const float*)d_in[5];
  const float* b_ih   = (const float*)d_in[6];
  const float* b_hh   = (const float*)d_in[7];
  const float* W1     = (const float*)d_in[8];
  const float* b1     = (const float*)d_in[9];
  const float* W2     = (const float*)d_in[10];
  const float* b2     = (const float*)d_in[11];
  const float* W3     = (const float*)d_in[12];
  const float* b3     = (const float*)d_in[13];
  const float* alphas = (const float*)d_in[14];
  char* ws = (char*)d_ws;
  float* out = (float*)d_out;

  k_prep<<<64, 256, 0, stream>>>(W_ih, W_hh, W1, W2, ws);
  k_lstm<<<1, 256, 0, stream>>>(user_seq, emb, b_ih, b_hh, ws);
  k_aprime<<<NL, 256, 0, stream>>>(W1, b1, ws);
  k_score<<<NIT / 16, 256, 0, stream>>>(emb, W3, b2, b3, alphas, ws, out);
  k_loss<<<1, 64, 0, stream>>>(pos_item, neg_item, out);
}
